// DrawingSeq2SeqPredictor_2310692405468
// MI455X (gfx1250) — matrix operand order repaired and checked by reading, not compiled, not run on hardware
//
#include <hip/hip_runtime.h>

// ---------------------------------------------------------------------------
// Seq2seq LSTM predictor for MI455X (gfx1250, wave32, WMMA f16 -> f32)
//
// B=131072, T_in=128, T_out=50, H=64, I=2.
// Each wave handles a 16-row batch tile; gates(16,256) = h(16,64) @ Whh^T
// computed as 32 v_wmma_f32_16x16x32_f16 per step. h/c state stays resident
// in VGPRs/LDS for all 178 steps; Whh staged into LDS as f16 once per phase.
// ---------------------------------------------------------------------------

typedef __attribute__((ext_vector_type(16))) _Float16 v16h;
typedef __attribute__((ext_vector_type(8)))  _Float16 v8h;
typedef __attribute__((ext_vector_type(8)))  float    v8f;

#define HDIM   64
#define T_IN   128
#define T_OUTN 50
#define G4     256   // 4*H

__device__ __forceinline__ float fast_sigmoid(float x) {
    return 1.0f / (1.0f + __expf(-x));
}
__device__ __forceinline__ float fast_tanh(float x) {
    return 2.0f / (1.0f + __expf(-2.0f * x)) - 1.0f;
}
__device__ __forceinline__ v16h cat8(v8h lo, v8h hi) {
    return __builtin_shufflevector(lo, hi, 0,1,2,3,4,5,6,7,8,9,10,11,12,13,14,15);
}

struct LaneConst {
    float bsum[16];  // bih[gc]+bhh[gc] for tile t: gc = 16*t + (lane&15)
    float w0[16];    // Wih[gc][0]
    float w1[16];    // Wih[gc][1]
};

__device__ __forceinline__ void load_consts(LaneConst& lc,
                                            const float* __restrict__ Wih,
                                            const float* __restrict__ bih,
                                            const float* __restrict__ bhh,
                                            int l16) {
#pragma unroll
    for (int t = 0; t < 16; ++t) {
        int gc = t * 16 + l16;
        lc.w0[t]   = Wih[gc * 2 + 0];
        lc.w1[t]   = Wih[gc * 2 + 1];
        lc.bsum[t] = bih[gc] + bhh[gc];
    }
}

// One LSTM cell step for this wave's 16-row batch tile.
//   hA:   A-matrix fragments of current h (2 chunks of K=32)
//   c:    cell state, lane-local (tile n in 0..3, reg r in 0..7)
//   sWhh: f16 weights [256][64] in LDS (row = gate col, col = hidden j)
//   sH:   per-wave h output buffer [16][64] f16 (row-major) -> next A frags
//   sX:   per-wave input [16] float2
__device__ __forceinline__ void lstm_step(const v16h hA[2], float c[4][8],
                                          const _Float16* __restrict__ sWhh,
                                          _Float16* __restrict__ sH,
                                          const float2* __restrict__ sX,
                                          const LaneConst& lc, int lane) {
    const int l16 = lane & 15;
    const int hi  = (lane >> 4) & 1;   // which half-wave
#pragma unroll
    for (int n = 0; n < 4; ++n) {      // hidden block j in [16n, 16n+16)
        v8f acc[4];
#pragma unroll
        for (int g = 0; g < 4; ++g) {  // gate: 0=i 1=f 2=g 3=o
            const int tile = g * 4 + n;          // N-tile index (0..15)
            const int gc   = tile * 16 + l16;    // gate column this lane owns
            v8f a = {};
#pragma unroll
            for (int kc = 0; kc < 2; ++kc) {
                // B fragment: lane l holds N=l%16, K = kc*32 + (hi?8:0) + {0..7, 16..23}
                const int j0 = kc * 32 + (hi ? 8 : 0);
                const _Float16* bp = sWhh + gc * HDIM + j0;
                v8h blo = *(const v8h*)(bp);
                v8h bhi = *(const v8h*)(bp + 16);
                v16h bfrag = cat8(blo, bhi);
                a = __builtin_amdgcn_wmma_f32_16x16x32_f16(
                        false, hA[kc], false, bfrag, (short)0, a, false, false);
            }
            acc[g] = a;
        }
        // Elementwise LSTM math: C/D layout => lane holds (row=r+8*hi, j=16n+l16)
#pragma unroll
        for (int r = 0; r < 8; ++r) {
            const int row = r + hi * 8;
            const float x0 = sX[row].x, x1 = sX[row].y;
            float gi = acc[0][r] + lc.bsum[n]      + x0 * lc.w0[n]      + x1 * lc.w1[n];
            float gf = acc[1][r] + lc.bsum[4 + n]  + x0 * lc.w0[4 + n]  + x1 * lc.w1[4 + n];
            float gg = acc[2][r] + lc.bsum[8 + n]  + x0 * lc.w0[8 + n]  + x1 * lc.w1[8 + n];
            float go = acc[3][r] + lc.bsum[12 + n] + x0 * lc.w0[12 + n] + x1 * lc.w1[12 + n];
            gi = fast_sigmoid(gi);
            gf = fast_sigmoid(gf);
            gg = fast_tanh(gg);
            go = fast_sigmoid(go);
            const float cv = gf * c[n][r] + gi * gg;
            c[n][r] = cv;
            const float hv = go * fast_tanh(cv);
            sH[row * HDIM + n * 16 + l16] = (_Float16)hv;
        }
    }
}

// Reload A-matrix fragments from the (row, j) h buffer.
// 16-bit A 16x32 layout: lanes 0-15: K 0..7 & 16..23 ; lanes 16-31: K 8..15 & 24..31
__device__ __forceinline__ void load_hA(v16h hA[2],
                                        const _Float16* __restrict__ sH,
                                        int lane) {
    const int m  = lane & 15;
    const int hi = (lane >> 4) & 1;
#pragma unroll
    for (int kc = 0; kc < 2; ++kc) {
        const int jb = kc * 32 + hi * 8;
        const _Float16* ap = sH + m * HDIM + jb;
        v8h alo = *(const v8h*)(ap);        // K = jb .. jb+7
        v8h ahi = *(const v8h*)(ap + 16);   // K = jb+16 .. jb+23
        hA[kc] = cat8(alo, ahi);
    }
}

__global__ __launch_bounds__(128) void seq2seq_lstm_kernel(
    const float* __restrict__ input_seq,
    const float* __restrict__ Wih_enc, const float* __restrict__ Whh_enc,
    const float* __restrict__ bih_enc, const float* __restrict__ bhh_enc,
    const float* __restrict__ Wih_dec, const float* __restrict__ Whh_dec,
    const float* __restrict__ bih_dec, const float* __restrict__ bhh_dec,
    const float* __restrict__ Wfc, const float* __restrict__ bfc,
    float* __restrict__ out) {
    __shared__ __align__(16) _Float16 sWhh[G4 * HDIM];   // 32 KB
    __shared__ __align__(16) _Float16 sH[4][16 * HDIM];  // 8 KB (per-wave h)
    __shared__ __align__(16) float2   sX[4][16];         // per-wave x
    __shared__ __align__(16) _Float16 sWfc[2 * HDIM];

    const int tid  = threadIdx.x;
    const int wid  = tid >> 5;
    const int lane = tid & 31;
    const int l16  = lane & 15;
    const long baseRow = (long)blockIdx.x * 64 + (long)wid * 16;

    // Stage encoder Whh (f32 -> f16) and Wfc into LDS.
    for (int i = tid; i < G4 * HDIM; i += 128) sWhh[i] = (_Float16)Whh_enc[i];
    for (int i = tid; i < 2 * HDIM; i += 128)  sWfc[i] = (_Float16)Wfc[i];

    LaneConst lc;
    load_consts(lc, Wih_enc, bih_enc, bhh_enc, l16);

    float c[4][8];
#pragma unroll
    for (int n = 0; n < 4; ++n)
#pragma unroll
        for (int r = 0; r < 8; ++r) c[n][r] = 0.0f;
    v16h hA[2] = {};  // h0 = 0

    __syncthreads();

    // ----------------- encoder: 128 steps -----------------
    for (int t = 0; t < T_IN; ++t) {
        if (lane < 16) {
            sX[wid][lane] = *(const float2*)(input_seq +
                              (baseRow + lane) * (long)(T_IN * 2) + t * 2);
        }
        __syncthreads();
        lstm_step(hA, c, sWhh, sH[wid], sX[wid], lc, lane);
        __syncthreads();
        load_hA(hA, sH[wid], lane);
        __syncthreads();
    }

    // ----------------- switch to decoder weights -----------------
    for (int i = tid; i < G4 * HDIM; i += 128) sWhh[i] = (_Float16)Whh_dec[i];
    load_consts(lc, Wih_dec, bih_dec, bhh_dec, l16);
    if (lane < 16) {   // dec_in0 = input_seq[:, -1, :]
        sX[wid][lane] = *(const float2*)(input_seq +
                          (baseRow + lane) * (long)(T_IN * 2) + (T_IN - 1) * 2);
    }
    __syncthreads();

    const float bfc0 = bfc[0];
    const float bfc1 = bfc[1];

    // ----------------- decoder: 50 steps + FC feedback -----------------
    for (int t = 0; t < T_OUTN; ++t) {
        lstm_step(hA, c, sWhh, sH[wid], sX[wid], lc, lane);
        __syncthreads();
        load_hA(hA, sH[wid], lane);
        // FC head: lane -> (row = lane&15, oc = lane>>4); point = sigmoid(h.Wfc^T+b)
        {
            const int row = l16;
            const int oc  = lane >> 4;
            float sum = (oc == 0) ? bfc0 : bfc1;
            const _Float16* hp = &sH[wid][row * HDIM];
            const _Float16* wp = &sWfc[oc * HDIM];
#pragma unroll
            for (int jc = 0; jc < 8; ++jc) {
                v8h hv = *(const v8h*)(hp + jc * 8);
                v8h wv = *(const v8h*)(wp + jc * 8);
#pragma unroll
                for (int e = 0; e < 8; ++e)
                    sum += (float)hv[e] * (float)wv[e];
            }
            const float p = fast_sigmoid(sum);
            out[(baseRow + row) * (long)(T_OUTN * 2) + t * 2 + oc] = p;
            ((float*)&sX[wid][row])[oc] = p;  // feed back as next x
        }
        __syncthreads();
    }
}

extern "C" void kernel_launch(void* const* d_in, const int* in_sizes, int n_in,
                              void* d_out, int out_size, void* d_ws, size_t ws_size,
                              hipStream_t stream) {
    (void)n_in; (void)out_size; (void)d_ws; (void)ws_size;
    const float* input_seq = (const float*)d_in[0];
    const float* Wih_enc   = (const float*)d_in[1];
    const float* Whh_enc   = (const float*)d_in[2];
    const float* bih_enc   = (const float*)d_in[3];
    const float* bhh_enc   = (const float*)d_in[4];
    const float* Wih_dec   = (const float*)d_in[5];
    const float* Whh_dec   = (const float*)d_in[6];
    const float* bih_dec   = (const float*)d_in[7];
    const float* bhh_dec   = (const float*)d_in[8];
    const float* Wfc       = (const float*)d_in[9];
    const float* bfc       = (const float*)d_in[10];
    float* out             = (float*)d_out;

    const int B = in_sizes[0] / (T_IN * 2);   // 131072
    const int blocks = B / 64;                // 64 batch rows per 128-thread block
    seq2seq_lstm_kernel<<<dim3(blocks), dim3(128), 0, stream>>>(
        input_seq, Wih_enc, Whh_enc, bih_enc, bhh_enc,
        Wih_dec, Whh_dec, bih_dec, bhh_dec, Wfc, bfc, out);
}
